// AngularPenaltySMLoss_32478542692386
// MI455X (gfx1250) — matrix operand order repaired and checked by reading, not compiled, not run on hardware
//
#include <hip/hip_runtime.h>

// ---------------------------------------------------------------------------
// ArcFace loss, fused for MI455X (gfx1250, wave32, WMMA bf16 + async-LDS).
//   x [1024,512] f32, W [100000,512] f32, labels [1024] i32 -> scalar f32
//
// Main kernel per 128-thread block (64 classes):
//   - stage normalized bf16 W chunk in LDS (read W from HBM exactly once)
//   - B operand (16 classes x K=512) register-resident: 128 VGPRs
//   - A tiles (32 rows x 512 bf16) double-buffered in LDS via
//     global_load_async_to_lds_b128 (ASYNCcnt), consumed with ds_load_b128
//   - A fragments software-pipelined one k-step ahead in registers so each
//     WMMA pair overlaps the next iteration's LDS reads (dscnt > 0 waits)
//   - fused epilogue: exp(s*cos), shuffle-reduce, global atomic row sums,
//     target-cosine capture
// ---------------------------------------------------------------------------

typedef __attribute__((ext_vector_type(16))) __bf16 v16bf;
typedef __attribute__((ext_vector_type(8)))  float  v8f;

#define N_ROWS   1024
#define DDIM     512
#define NCLS     100000
#define S_SCALE  64.0f
#define MARGIN   0.5f
#define EPS_C    1e-7f

#define CHUNK       64     // classes per block
#define ROW_TILE    32     // x-rows per main-loop iteration (2 WMMA chains)
#define NT          (N_ROWS / ROW_TILE)            // 32 iterations
#define LDS_STRIDE  520    // row stride in bf16 elems (1040 B = 260 dwords, %64==4)
#define LDSW_ELEMS  (CHUNK * LDS_STRIDE)           // 33280 elems = 66560 B
#define LDSA_ELEMS  (ROW_TILE * LDS_STRIDE)        // 16640 elems = 33280 B
#define LDS_TOTAL_B ((LDSW_ELEMS + 2 * LDSA_ELEMS) * 2)   // 133120 bytes

// ws layout (bytes):
#define WS_XN    0          // xn: 1024*512 bf16 (1 MB)
#define WS_INVW  1048576    // invW: 100000 f32
#define WS_RSUM  1448576    // rowsum: 1024 f32 (zeroed each call)
#define WS_TCOS  1452672    // tcos:   1024 f32 (zeroed each call)

union V16 { v16bf v; uint4 u[2]; };

// --- K1: zero the per-row accumulators (rowsum + tcos, contiguous) ----------
__global__ void arc_init_kernel(float* __restrict__ acc) {
    int i = blockIdx.x * blockDim.x + threadIdx.x;   // 2048 threads
    acc[i] = 0.0f;
}

// --- K2: L2-normalize x rows -> bf16. One wave32 per row. -------------------
__global__ void arc_norm_x_kernel(const float* __restrict__ x,
                                  __bf16* __restrict__ xn) {
    int wave = (blockIdx.x * blockDim.x + threadIdx.x) >> 5;   // 0..1023
    int lane = threadIdx.x & 31;
    const float* r = x + (size_t)wave * DDIM;
    float v[16];
    float ss = 0.0f;
    #pragma unroll
    for (int j = 0; j < 16; ++j) { v[j] = r[lane + 32 * j]; ss += v[j] * v[j]; }
    #pragma unroll
    for (int m = 16; m; m >>= 1) ss += __shfl_xor(ss, m, 32);
    float inv = rsqrtf(ss);
    __bf16* o = xn + (size_t)wave * DDIM;
    #pragma unroll
    for (int j = 0; j < 16; ++j) o[lane + 32 * j] = (__bf16)(v[j] * inv);
}

// --- K3: 1/||W_row||. One wave32 per row (grid sized exactly). --------------
__global__ void arc_winv_kernel(const float* __restrict__ W,
                                float* __restrict__ invW) {
    int wave = (blockIdx.x * blockDim.x + threadIdx.x) >> 5;   // 0..99999
    int lane = threadIdx.x & 31;
    const float* r = W + (size_t)wave * DDIM;
    float ss = 0.0f;
    #pragma unroll
    for (int j = 0; j < 16; ++j) { float v = r[lane + 32 * j]; ss += v * v; }
    #pragma unroll
    for (int m = 16; m; m >>= 1) ss += __shfl_xor(ss, m, 32);
    if (lane == 0) invW[wave] = rsqrtf(ss);
}

// Issue the async global->LDS copy of one 32-row x 1KB A tile.
// 128 lanes x 16 B x 16 issues = 32 KB (row-padded destination).
__device__ __forceinline__ void issue_tile_async(unsigned ldsBufByte,
                                                 const __bf16* xn, int t,
                                                 int tid) {
    const unsigned chunk   = (unsigned)(tid & 63) * 16;   // byte offset in row
    const unsigned rowPair = (unsigned)(tid >> 6);        // 0 or 1
    const unsigned long long gbase =
        (unsigned long long)(uintptr_t)xn + (unsigned long long)t * (ROW_TILE * DDIM * 2);
    #pragma unroll
    for (int j = 0; j < 16; ++j) {
        unsigned row = (unsigned)j * 2 + rowPair;
        unsigned lds = ldsBufByte + row * (LDS_STRIDE * 2) + chunk;
        unsigned gof = row * (DDIM * 2) + chunk;
        asm volatile("global_load_async_to_lds_b128 %0, %1, %2"
                     :
                     : "v"(lds), "v"(gof), "s"(gbase)
                     : "memory");
    }
}

// --- K4: fused GEMM (WMMA bf16) + exp/row-sum/target epilogue ---------------
__global__ void __launch_bounds__(128, 1)
arc_main_kernel(const __bf16* __restrict__ xn,
                const float*  __restrict__ W,
                const float*  __restrict__ invW,
                const int*    __restrict__ labels,
                float* __restrict__ rowsum,
                float* __restrict__ tcos) {
    extern __shared__ __bf16 ldsW[];   // [W chunk | A buf0 | A buf1]
    const int classBase = blockIdx.x * CHUNK;
    const int tid = threadIdx.x;

    __bf16* aBuf0 = ldsW + LDSW_ELEMS;
    __bf16* aBuf1 = aBuf0 + LDSA_ELEMS;
    // Low 32 bits of a generic LDS pointer == LDS byte offset (ISA 10.2).
    const unsigned aBuf0B = (unsigned)(uintptr_t)aBuf0;
    const unsigned aBuf1B = (unsigned)(uintptr_t)aBuf1;

    // Kick off tile 0 async copy; it overlaps the W-chunk staging below.
    issue_tile_async(aBuf0B, xn, 0, tid);

    // Stage: load W rows (f32), scale by invnorm, convert to bf16 in LDS.
    for (int i = tid * 4; i < CHUNK * DDIM; i += 128 * 4) {
        int row = i >> 9;            // /512
        int col = i & (DDIM - 1);
        int gc  = classBase + row;
        float4 f = make_float4(0.f, 0.f, 0.f, 0.f);
        float inv = 0.0f;
        if (gc < NCLS) {
            f   = *(const float4*)(W + (size_t)gc * DDIM + col);
            inv = invW[gc];
        }
        __bf16* dst = ldsW + row * LDS_STRIDE + col;
        dst[0] = (__bf16)(f.x * inv);
        dst[1] = (__bf16)(f.y * inv);
        dst[2] = (__bf16)(f.z * inv);
        dst[3] = (__bf16)(f.w * inv);
    }
    __syncthreads();

    const int  lane = tid & 31;
    const int  wave = tid >> 5;
    const bool hiH  = (lane >= 16);
    const int  nL   = wave * 16 + (lane & 15);     // class within chunk
    const int  cls  = classBase + nL;
    const bool clsOK = (cls < NCLS);
    const __bf16* bBase = ldsW + nL * LDS_STRIDE + (hiH ? 8 : 0);

    // Register-resident B operand: 16 fragments * 8 VGPRs = 128 VGPRs.
    V16 bfrag[16];
    #pragma unroll
    for (int k = 0; k < 16; ++k) {
        const uint4* bp = (const uint4*)(bBase + k * 32);
        bfrag[k].u[0] = bp[0];
        bfrag[k].u[1] = bp[2];
    }

    const int aOff = (hiH ? 8 : 0);            // elems
    const int mA   = (lane & 15);

    for (int t = 0; t < NT; ++t) {
        // Prefetch next tile into the other buffer, then wait for this tile.
        if (t + 1 < NT) {
            issue_tile_async(((t + 1) & 1) ? aBuf1B : aBuf0B, xn, t + 1, tid);
            asm volatile("s_wait_asynccnt 16" ::: "memory");
        } else {
            asm volatile("s_wait_asynccnt 0" ::: "memory");
        }
        __syncthreads();   // all waves' portions of tile t visible

        const __bf16* aBuf = (t & 1) ? aBuf1 : aBuf0;
        const __bf16* aRow0 = aBuf + mA * LDS_STRIDE + aOff;
        const __bf16* aRow1 = aRow0 + 16 * LDS_STRIDE;
        v8f acc0 = {};
        v8f acc1 = {};

        // Software-pipelined A fragments: preload k+1 while WMMA consumes k.
        V16 a0c, a1c, a0n, a1n;
        a0c.u[0] = *(const uint4*)(aRow0);
        a0c.u[1] = *(const uint4*)(aRow0 + 16);
        a1c.u[0] = *(const uint4*)(aRow1);
        a1c.u[1] = *(const uint4*)(aRow1 + 16);
        #pragma unroll
        for (int k = 0; k < 16; ++k) {
            if (k + 1 < 16) {
                a0n.u[0] = *(const uint4*)(aRow0 + (k + 1) * 32);
                a0n.u[1] = *(const uint4*)(aRow0 + (k + 1) * 32 + 16);
                a1n.u[0] = *(const uint4*)(aRow1 + (k + 1) * 32);
                a1n.u[1] = *(const uint4*)(aRow1 + (k + 1) * 32 + 16);
            }
            acc0 = __builtin_amdgcn_wmma_f32_16x16x32_bf16(
                       false, a0c.v, false, bfrag[k].v, (short)0, acc0, false, false);
            acc1 = __builtin_amdgcn_wmma_f32_16x16x32_bf16(
                       false, a1c.v, false, bfrag[k].v, (short)0, acc1, false, false);
            a0c = a0n;
            a1c = a1n;
        }

        // Epilogue: accX[r] = cos(row, class cls); rows differ per half-wave.
        #pragma unroll
        for (int r = 0; r < 8; ++r) {
            int   row0 = t * ROW_TILE + r + (hiH ? 8 : 0);
            int   row1 = row0 + 16;
            float cv0  = acc0[r];
            float cv1  = acc1[r];
            float e0   = clsOK ? __expf(S_SCALE * cv0) : 0.0f;
            float e1   = clsOK ? __expf(S_SCALE * cv1) : 0.0f;
            if (clsOK && labels[row0] == cls) tcos[row0] = cv0;
            if (clsOK && labels[row1] == cls) tcos[row1] = cv1;
            e0 += __shfl_xor(e0, 1, 32);  e1 += __shfl_xor(e1, 1, 32);
            e0 += __shfl_xor(e0, 2, 32);  e1 += __shfl_xor(e1, 2, 32);
            e0 += __shfl_xor(e0, 4, 32);  e1 += __shfl_xor(e1, 4, 32);
            e0 += __shfl_xor(e0, 8, 32);  e1 += __shfl_xor(e1, 8, 32);
            if ((lane & 15) == 0) {
                atomicAdd(&rowsum[row0], e0);
                atomicAdd(&rowsum[row1], e1);
            }
        }
        __syncthreads();   // done reading this buffer before it is refilled
    }
}

// --- K5: arcface finish + mean reduction ------------------------------------
__global__ void arc_final_kernel(const float* __restrict__ rowsum,
                                 const float* __restrict__ tcos,
                                 float* __restrict__ out) {
    __shared__ float part[32];
    int i = threadIdx.x;                  // 1024 threads
    float tc   = tcos[i];
    float tcc  = fminf(fmaxf(tc, -1.0f + EPS_C), 1.0f - EPS_C);
    float num  = S_SCALE * cosf(acosf(tcc) + MARGIN);
    float excl = fmaxf(rowsum[i] - __expf(S_SCALE * tc), 0.0f);
    float L    = num - logf(__expf(num) + excl);
    #pragma unroll
    for (int m = 16; m; m >>= 1) L += __shfl_xor(L, m, 32);
    if ((i & 31) == 0) part[i >> 5] = L;
    __syncthreads();
    if (i < 32) {
        float v = part[i];
        #pragma unroll
        for (int m = 16; m; m >>= 1) v += __shfl_xor(v, m, 32);
        if (i == 0) *out = -v / (float)N_ROWS;
    }
}

extern "C" void kernel_launch(void* const* d_in, const int* in_sizes, int n_in,
                              void* d_out, int out_size, void* d_ws, size_t ws_size,
                              hipStream_t stream) {
    (void)in_sizes; (void)n_in; (void)out_size; (void)ws_size;
    const float* x      = (const float*)d_in[0];
    const float* W      = (const float*)d_in[1];
    const int*   labels = (const int*)d_in[2];
    float* out = (float*)d_out;

    char*   ws     = (char*)d_ws;
    __bf16* xn     = (__bf16*)(ws + WS_XN);
    float*  invW   = (float*)(ws + WS_INVW);
    float*  rowsum = (float*)(ws + WS_RSUM);
    float*  tcosv  = (float*)(ws + WS_TCOS);

    // K1: zero rowsum + tcos (2048 contiguous floats)
    arc_init_kernel<<<8, 256, 0, stream>>>(rowsum);
    // K2: normalize x -> bf16 (1024 rows, wave per row, 8 waves/block)
    arc_norm_x_kernel<<<N_ROWS / 8, 256, 0, stream>>>(x, xn);
    // K3: W row inverse norms (100000 rows, wave per row)
    arc_winv_kernel<<<NCLS / 8, 256, 0, stream>>>(W, invW);
    // K4: fused WMMA GEMM + async A pipeline + softmax-sum epilogue
    int nChunks = (NCLS + CHUNK - 1) / CHUNK;                 // 1563
    arc_main_kernel<<<nChunks, 128, LDS_TOTAL_B, stream>>>(xn, W, invW, labels,
                                                           rowsum, tcosv);
    // K5: finish
    arc_final_kernel<<<1, N_ROWS, 0, stream>>>(rowsum, tcosv, out);
}
